// LSTMCell_42855183680049
// MI455X (gfx1250) — matrix operand order repaired and checked by reading, not compiled, not run on hardware
//
#include <hip/hip_runtime.h>
#include <math.h>

typedef __attribute__((ext_vector_type(16))) __bf16 v16bf;
typedef __attribute__((ext_vector_type(8)))  float  v8f;

namespace {
constexpr int BATCH = 4096;
constexpr int KDIM  = 2048;   // I == H
constexpr int HDIM  = 2048;
constexpr int GDIM  = 8192;   // 4*H
}

__device__ __forceinline__ unsigned short f32_to_bf16(float f) {
    unsigned int u = __float_as_uint(f);
    u += 0x7FFFu + ((u >> 16) & 1u);       // round-to-nearest-even
    return (unsigned short)(u >> 16);
}

// ---------------------------------------------------------------------------
// 1) elementwise f32 -> bf16
// ---------------------------------------------------------------------------
__global__ void __launch_bounds__(256)
cvt_bf16_kernel(const float* __restrict__ in, unsigned short* __restrict__ out) {
    int base = blockIdx.x * 1024 + threadIdx.x;
#pragma unroll
    for (int u = 0; u < 4; ++u) {
        int idx = base + u * 256;
        out[idx] = f32_to_bf16(in[idx]);
    }
}

// ---------------------------------------------------------------------------
// 2) transpose + convert: in f32 [rows, cols] -> out bf16 [cols, rows]
// ---------------------------------------------------------------------------
__global__ void __launch_bounds__(256)
transpose_cvt_kernel(const float* __restrict__ in, unsigned short* __restrict__ out,
                     int rows, int cols) {
    __shared__ float tile[32][33];
    int cx = blockIdx.x * 32 + threadIdx.x;      // input col
    int ry = blockIdx.y * 32 + threadIdx.y;      // input row
#pragma unroll
    for (int i = 0; i < 32; i += 8)
        tile[threadIdx.y + i][threadIdx.x] = in[(size_t)(ry + i) * cols + cx];
    __syncthreads();
    int ox = blockIdx.y * 32 + threadIdx.x;      // output col (= input row)
    int oy = blockIdx.x * 32 + threadIdx.y;      // output row (= input col)
#pragma unroll
    for (int i = 0; i < 32; i += 8)
        out[(size_t)(oy + i) * rows + ox] = f32_to_bf16(tile[threadIdx.x][threadIdx.y + i]);
}

// ---------------------------------------------------------------------------
// 3) dual bf16 WMMA GEMM: C[M,N] = A[M,K] * Bt[N,K]^T + bias[N]
//    Block tile 128x256x32, 8 wave32, 64x64 per wave -> 16 WMMA / K-step.
//    Software-pipelined: next tile's global loads issued before WMMA burst.
// ---------------------------------------------------------------------------
__global__ void __launch_bounds__(256)
gemm_dual_bf16_wmma(const unsigned short* __restrict__ A0,
                    const unsigned short* __restrict__ Bt0,
                    const float* __restrict__ bias0, float* __restrict__ C0,
                    const unsigned short* __restrict__ A1,
                    const unsigned short* __restrict__ Bt1,
                    const float* __restrict__ bias1, float* __restrict__ C1,
                    int K, int N) {
    constexpr int LDT = 40;                      // padded LDS row stride (elems)
    __shared__ unsigned short As[128 * LDT];     // 10240 B
    __shared__ unsigned short Bs[256 * LDT];     // 20480 B

    const unsigned short* A    = blockIdx.z ? A1    : A0;
    const unsigned short* Bt   = blockIdx.z ? Bt1   : Bt0;
    const float*          bias = blockIdx.z ? bias1 : bias0;
    float*                C    = blockIdx.z ? C1    : C0;

    const int tid   = threadIdx.x;
    const int lane  = tid & 31;
    const int wave  = tid >> 5;        // 0..7
    const int wm    = wave & 1;        // 2 slabs of 64 rows
    const int wn    = wave >> 1;       // 4 slabs of 64 cols
    const int l16   = lane & 15;
    const int khalf = lane >> 4;

    const int rowBase = blockIdx.y * 128;
    const int colBase = blockIdx.x * 256;

    v8f acc[4][4];
#pragma unroll
    for (int mt = 0; mt < 4; ++mt)
#pragma unroll
        for (int nt = 0; nt < 4; ++nt)
            acc[mt][nt] = (v8f){};

    // A staging: 2 threads per tile row (128 rows x 32 elems)
    const int aRow = tid >> 1;
    const int aSeg = (tid & 1) * 16;
    const unsigned short* aPtr = A + (size_t)(rowBase + aRow) * K + aSeg;
    // B staging: 1 thread per tile row (256 rows x 32 elems)
    const unsigned short* bPtr = Bt + (size_t)(colBase + tid) * K;

    // ---- prologue: fetch tile 0 into registers
    uint4 ra0 = *(const uint4*)(aPtr);
    uint4 ra1 = *(const uint4*)(aPtr + 8);
    uint4 rb0 = *(const uint4*)(bPtr);
    uint4 rb1 = *(const uint4*)(bPtr + 8);
    uint4 rb2 = *(const uint4*)(bPtr + 16);
    uint4 rb3 = *(const uint4*)(bPtr + 24);

    for (int k0 = 0; k0 < K; k0 += 32) {
        __syncthreads();                       // prior reads of LDS done
        *(uint4*)(As + aRow * LDT + aSeg)     = ra0;
        *(uint4*)(As + aRow * LDT + aSeg + 8) = ra1;
        *(uint4*)(Bs + tid * LDT)             = rb0;
        *(uint4*)(Bs + tid * LDT + 8)         = rb1;
        *(uint4*)(Bs + tid * LDT + 16)        = rb2;
        *(uint4*)(Bs + tid * LDT + 24)        = rb3;
        __syncthreads();                       // tile visible

        // ---- issue NEXT tile's global loads now; they overlap the WMMA burst
        if (k0 + 32 < K) {
            ra0 = *(const uint4*)(aPtr + k0 + 32);
            ra1 = *(const uint4*)(aPtr + k0 + 40);
            rb0 = *(const uint4*)(bPtr + k0 + 32);
            rb1 = *(const uint4*)(bPtr + k0 + 40);
            rb2 = *(const uint4*)(bPtr + k0 + 48);
            rb3 = *(const uint4*)(bPtr + k0 + 56);
            if (k0 + 64 < K) {
                __builtin_prefetch((const void*)(aPtr + k0 + 64), 0, 3);
                __builtin_prefetch((const void*)(bPtr + k0 + 64), 0, 3);
            }
        }

        // ---- batch ALL fragment loads, then an uninterrupted 16-WMMA burst
        union U { uint4 u[2]; v16bf v; };
        // B frag: lane l16 = column n, VGPRs hold K = khalf*16 + 0..15
        U bf[4];
#pragma unroll
        for (int nt = 0; nt < 4; ++nt) {
            int n = wn * 64 + nt * 16 + l16;
            const uint4* p = (const uint4*)(Bs + n * LDT + khalf * 16);
            bf[nt].u[0] = p[0];
            bf[nt].u[1] = p[1];
        }
        // A frag: lane l16 = row m, VGPRs hold K = khalf*16 + 0..15
        U af[4];
#pragma unroll
        for (int mt = 0; mt < 4; ++mt) {
            int r = wm * 64 + mt * 16 + l16;
            af[mt].u[0] = *(const uint4*)(As + r * LDT + khalf * 16);
            af[mt].u[1] = *(const uint4*)(As + r * LDT + khalf * 16 + 8);
        }
        __builtin_amdgcn_sched_barrier(0);     // loads above, WMMA burst below
#pragma unroll
        for (int mt = 0; mt < 4; ++mt)
#pragma unroll
            for (int nt = 0; nt < 4; ++nt)
                acc[mt][nt] = __builtin_amdgcn_wmma_f32_16x16x32_bf16(
                    false, af[mt].v, false, bf[nt].v, (short)0, acc[mt][nt], false, false);
    }

    // epilogue: C/D layout — VGPR r: M = r (+8 for lanes 16-31), N = lane&15
    const int mrow0 = (lane < 16) ? 0 : 8;
#pragma unroll
    for (int mt = 0; mt < 4; ++mt)
#pragma unroll
        for (int nt = 0; nt < 4; ++nt) {
            int col   = colBase + wn * 64 + nt * 16 + l16;
            float bv  = bias[col];
            size_t bs = (size_t)(rowBase + wm * 64 + mt * 16 + mrow0) * N + col;
#pragma unroll
            for (int r = 0; r < 8; ++r)
                C[bs + (size_t)r * N] = acc[mt][nt][r] + bv;
        }
}

// ---------------------------------------------------------------------------
// 4) per-row mean / rstd over ncols
// ---------------------------------------------------------------------------
__global__ void __launch_bounds__(256)
rowstats_kernel(const float* __restrict__ X, int ncols,
                float* __restrict__ mu, float* __restrict__ rs) {
    __shared__ float s1[256], s2[256];
    int row = blockIdx.x;
    const float* p = X + (size_t)row * ncols;
    float a = 0.f, b = 0.f;
    for (int i = threadIdx.x; i < ncols; i += 256) {
        float v = p[i]; a += v; b += v * v;
    }
    s1[threadIdx.x] = a; s2[threadIdx.x] = b;
    __syncthreads();
    for (int off = 128; off > 0; off >>= 1) {
        if (threadIdx.x < off) {
            s1[threadIdx.x] += s1[threadIdx.x + off];
            s2[threadIdx.x] += s2[threadIdx.x + off];
        }
        __syncthreads();
    }
    if (threadIdx.x == 0) {
        float m   = s1[0] / (float)ncols;
        float var = s2[0] / (float)ncols - m * m;
        mu[row] = m;
        rs[row] = rsqrtf(var + 1e-5f);
    }
}

// ---------------------------------------------------------------------------
// 5) fused gate math + two LayerNorms; one block per batch row
// ---------------------------------------------------------------------------
__global__ void __launch_bounds__(256)
lstm_fused_kernel(const float* __restrict__ gx, const float* __restrict__ gh,
                  const float* __restrict__ mu1, const float* __restrict__ rs1,
                  const float* __restrict__ g_x, const float* __restrict__ beta_x,
                  const float* __restrict__ c_prev,
                  const float* __restrict__ g_c, const float* __restrict__ beta_c,
                  const float* __restrict__ g_h, const float* __restrict__ beta_h,
                  float* __restrict__ h_out, float* __restrict__ c_out) {
    constexpr int H = HDIM, G = GDIM;
    __shared__ float s1[256], s2[256];
    __shared__ float stats[2];

    int b = blockIdx.x;
    float m1 = mu1[b], r1 = rs1[b];
    const float* gxr = gx + (size_t)b * G;
    const float* ghr = gh + (size_t)b * G;

    float craw[8], og[8];
    float a = 0.f, sq = 0.f;
#pragma unroll
    for (int u = 0; u < 8; ++u) {
        int j = u * 256 + threadIdx.x;
        float vi = (gxr[j]         - m1) * r1 * g_x[j]         + beta_x[j]         + ghr[j];
        float vf = (gxr[H + j]     - m1) * r1 * g_x[H + j]     + beta_x[H + j]     + ghr[H + j];
        float vg = (gxr[2 * H + j] - m1) * r1 * g_x[2 * H + j] + beta_x[2 * H + j] + ghr[2 * H + j];
        float vo = (gxr[3 * H + j] - m1) * r1 * g_x[3 * H + j] + beta_x[3 * H + j] + ghr[3 * H + j];
        float it = 1.f / (1.f + __expf(-vi));
        float ft = 1.f / (1.f + __expf(-vf));
        float gt = tanhf(vg);
        float ot = 1.f / (1.f + __expf(-vo));
        float cr = ft * c_prev[(size_t)b * H + j] + it * gt;
        craw[u] = cr; og[u] = ot;
        a += cr; sq += cr * cr;
    }
    // reduce #1 -> c stats
    s1[threadIdx.x] = a; s2[threadIdx.x] = sq;
    __syncthreads();
    for (int off = 128; off > 0; off >>= 1) {
        if (threadIdx.x < off) {
            s1[threadIdx.x] += s1[threadIdx.x + off];
            s2[threadIdx.x] += s2[threadIdx.x + off];
        }
        __syncthreads();
    }
    if (threadIdx.x == 0) {
        float m = s1[0] / (float)H;
        stats[0] = m;
        stats[1] = rsqrtf(s2[0] / (float)H - m * m + 1e-5f);
    }
    __syncthreads();
    float mc = stats[0], rc = stats[1];

    float hraw[8];
    a = 0.f; sq = 0.f;
#pragma unroll
    for (int u = 0; u < 8; ++u) {
        int j = u * 256 + threadIdx.x;
        float ct = (craw[u] - mc) * rc * g_c[j] + beta_c[j];
        c_out[(size_t)b * H + j] = ct;
        float hr = og[u] * tanhf(ct);
        hraw[u] = hr;
        a += hr; sq += hr * hr;
    }
    // reduce #2 -> h stats
    s1[threadIdx.x] = a; s2[threadIdx.x] = sq;
    __syncthreads();
    for (int off = 128; off > 0; off >>= 1) {
        if (threadIdx.x < off) {
            s1[threadIdx.x] += s1[threadIdx.x + off];
            s2[threadIdx.x] += s2[threadIdx.x + off];
        }
        __syncthreads();
    }
    if (threadIdx.x == 0) {
        float m = s1[0] / (float)H;
        stats[0] = m;
        stats[1] = rsqrtf(s2[0] / (float)H - m * m + 1e-5f);
    }
    __syncthreads();
    float mh = stats[0], rh = stats[1];
#pragma unroll
    for (int u = 0; u < 8; ++u) {
        int j = u * 256 + threadIdx.x;
        h_out[(size_t)b * H + j] = (hraw[u] - mh) * rh * g_h[j] + beta_h[j];
    }
}

// ---------------------------------------------------------------------------
// launch
// ---------------------------------------------------------------------------
extern "C" void kernel_launch(void* const* d_in, const int* in_sizes, int n_in,
                              void* d_out, int out_size, void* d_ws, size_t ws_size,
                              hipStream_t stream) {
    (void)in_sizes; (void)n_in; (void)out_size; (void)ws_size;

    const float* x      = (const float*)d_in[0];
    const float* h_prev = (const float*)d_in[1];
    const float* c_prev = (const float*)d_in[2];
    const float* W_ih   = (const float*)d_in[3];
    const float* W_hh   = (const float*)d_in[4];
    const float* b_ih   = (const float*)d_in[5];
    const float* b_hh   = (const float*)d_in[6];
    const float* g_x    = (const float*)d_in[7];
    const float* beta_x = (const float*)d_in[8];
    const float* g_c    = (const float*)d_in[9];
    const float* beta_c = (const float*)d_in[10];
    const float* g_h    = (const float*)d_in[11];
    const float* beta_h = (const float*)d_in[12];

    char* ws = (char*)d_ws;
    size_t off = 0;
    unsigned short* x_bf  = (unsigned short*)(ws + off); off += (size_t)BATCH * KDIM * 2;
    unsigned short* h_bf  = (unsigned short*)(ws + off); off += (size_t)BATCH * KDIM * 2;
    unsigned short* Wt_ih = (unsigned short*)(ws + off); off += (size_t)GDIM * KDIM * 2;
    unsigned short* Wt_hh = (unsigned short*)(ws + off); off += (size_t)GDIM * KDIM * 2;
    float* gx  = (float*)(ws + off); off += (size_t)BATCH * GDIM * 4;
    float* gh  = (float*)(ws + off); off += (size_t)BATCH * GDIM * 4;
    float* mu1 = (float*)(ws + off); off += (size_t)BATCH * 4;
    float* rs1 = (float*)(ws + off); off += (size_t)BATCH * 4;

    float* h_out = (float*)d_out;
    float* c_out = h_out + (size_t)BATCH * HDIM;

    // 1) activations -> bf16
    cvt_bf16_kernel<<<(BATCH * KDIM) / 1024, 256, 0, stream>>>(x, x_bf);
    cvt_bf16_kernel<<<(BATCH * KDIM) / 1024, 256, 0, stream>>>(h_prev, h_bf);

    // 2) weights -> bf16 transposed [G, K]
    {
        dim3 blk(32, 8);
        dim3 grd(GDIM / 32, KDIM / 32);
        transpose_cvt_kernel<<<grd, blk, 0, stream>>>(W_ih, Wt_ih, KDIM, GDIM);
        transpose_cvt_kernel<<<grd, blk, 0, stream>>>(W_hh, Wt_hh, KDIM, GDIM);
    }

    // 3+4) gx = x @ W_ih + b_ih  and  gh = h_prev @ W_hh + b_hh  (one launch)
    {
        dim3 grd(GDIM / 256, BATCH / 128, 2);
        gemm_dual_bf16_wmma<<<grd, 256, 0, stream>>>(x_bf, Wt_ih, b_ih, gx,
                                                     h_bf, Wt_hh, b_hh, gh,
                                                     KDIM, GDIM);
    }
    // 5) row stats of gx (for LN over the G axis)
    rowstats_kernel<<<BATCH, 256, 0, stream>>>(gx, GDIM, mu1, rs1);
    // 6) fused gates + cell update + LN(c) + LN(h)
    lstm_fused_kernel<<<BATCH, 256, 0, stream>>>(gx, gh, mu1, rs1, g_x, beta_x,
                                                 c_prev, g_c, beta_c, g_h, beta_h,
                                                 h_out, c_out);
}
